// Spiking_vit_MetaFormer_Spike_SepConv_23313082483188
// MI455X (gfx1250) — matrix operand order repaired and checked by reading, not compiled, not run on hardware
//
#include <hip/hip_runtime.h>
#include <hip/hip_bf16.h>

// Spiking MetaFormer block on gfx1250 (MI455X): wave32, WMMA f16 matmuls,
// GLOBAL_LOAD_ASYNC_TO_LDS staging (ASYNCcnt), token-major spike layouts.
// All GEMM activations are binary spikes (exact in f16); attention kv is
// integer-valued <=576 (exact in f16) -> every matmul on v_wmma_f32_16x16x32_f16.

typedef _Float16 half_t;
typedef __attribute__((ext_vector_type(16))) _Float16 v16h;
typedef __attribute__((ext_vector_type(8)))  float    v8f;
typedef __attribute__((ext_vector_type(4)))  unsigned int u32x4;

union Frag { v16h v; u32x4 u[2]; };

// Async global->LDS staging (builtins confirmed by round-3/4 probes).
#if defined(__AMDGCN__) && __has_builtin(__builtin_amdgcn_global_load_async_to_lds_b128) && \
    __has_builtin(__builtin_amdgcn_s_wait_asynccnt)
#define USE_ASYNC 1
#else
#define USE_ASYNC 0
#endif

#if USE_ASYNC
// Round-4 diagnostic: parameter pointee type is gcc-style v4i (vector_size 16).
typedef int v4i_t __attribute__((vector_size(16)));
typedef __attribute__((address_space(1))) v4i_t as1_v4i;
typedef __attribute__((address_space(3))) v4i_t as3_v4i;

// Copy 32 bytes (one 16-half LDS row chunk) global->LDS, no VGPR round-trip.
// INST_OFFSET applies to both global and LDS addresses (ISA 08_async_tensor §4.4),
// so the two 16B chunks share base pointers with imm offsets 0/16.
__device__ __forceinline__ void stage32(const half_t* g, half_t* l)
{
    as1_v4i* gp = (as1_v4i*)(unsigned long long)(const void*)g;
    // low 32 bits of a generic LDS address are the LDS byte offset (aperture scheme)
    as3_v4i* lp = (as3_v4i*)(unsigned)(unsigned long long)(const void*)l;
    __builtin_amdgcn_global_load_async_to_lds_b128(gp, lp, 0, 0);
    __builtin_amdgcn_global_load_async_to_lds_b128(gp, lp, 16, 0);
}
#endif

// ---------------------------------------------------------------------------
// Batched GEMM:  C[z] = epi( alpha * A[z] x B[z] ), BM x 64 x K tiles.
//   A: row-major [M,K], ld=lda; per-z offset (z/zmod)*sAhi + (z%zmod)*sAlo
//   B: logical [K,N], K-contiguous per column: B[k,n] = Bsrc[n*ldb + k]
//   C: ct==0 -> C[m*ldc+n]; ct==1 -> C[n*ldc+m]
//   epilogue: v = alpha*acc; +bias2[m]; *scale[m]+bias[m];
//             +resid (rtm ? [n*ldr+m] : [m*ldr+n])
// 128 threads = 4 waves (2x2); per-wave tile = (BM/2) x 32; double-buffered
// LDS fed by GLOBAL_LOAD_ASYNC_TO_LDS_B128 (fallback: register staging).
// Requires M%BM==0, N%64==0, K%32==0, lda/ldb multiples of 8 halves.
// ---------------------------------------------------------------------------
template <int BM>
__global__ __launch_bounds__(128) void wmma_gemm(
    const half_t* __restrict__ A, long sAhi, long sAlo, int lda,
    const half_t* __restrict__ B, long sBhi, long sBlo, int ldb,
    float* __restrict__ C, long sChi, long sClo, int ldc, int ct,
    const float* __restrict__ bias2,
    const float* __restrict__ scale, const float* __restrict__ bias,
    const float* __restrict__ resid, long sRhi, long sRlo, int ldr, int rtm,
    int zmod, int M, int N, int K, float alpha)
{
    constexpr int MT = BM / 32;          // m-tiles per wave (2 or 4)
    constexpr int RH = BM / 64;          // A row-chunks per thread (1 or 2)
    __shared__ __align__(16) half_t As[2][BM][40];
    __shared__ __align__(16) half_t Bs[2][64][40];

    const int zh = blockIdx.z / zmod;
    const int zl = blockIdx.z - zh * zmod;
    A += zh * sAhi + zl * sAlo;
    B += zh * sBhi + zl * sBlo;
    C += zh * sChi + zl * sClo;
    if (resid) resid += zh * sRhi + zl * sRlo;

    const int m0 = blockIdx.y * BM;
    const int n0 = blockIdx.x * 64;
    const int tid = threadIdx.x;
    const int lane = tid & 31;
    const int wave = tid >> 5;
    const int wm = (wave & 1) * (BM / 2);
    const int wn = (wave >> 1) * 32;

    const int sr = tid >> 1;             // staging row 0..63
    const int sk = (tid & 1) * 16;       // staging k-offset (halves)
    const half_t* gA0 = A + (long)(m0 + sr) * lda + sk;
    const half_t* gA1 = A + (long)(m0 + sr + 64) * lda + sk;  // used iff RH==2
    const half_t* gB  = B + (long)(n0 + sr) * ldb + sk;

    v8f acc[MT][2] = {};
    const int nk = K >> 5;

#if USE_ASYNC
    auto stage_tile = [&](long ko, int bufi) {
        stage32(gA0 + ko, &As[bufi][sr][sk]);
        if constexpr (RH == 2) stage32(gA1 + ko, &As[bufi][sr + 64][sk]);
        stage32(gB + ko, &Bs[bufi][sr][sk]);
    };
    stage_tile(0, 0);
    __builtin_amdgcn_s_wait_asynccnt(0);
    __syncthreads();
#else
    u32x4 ra0[RH], ra1[RH], rb0, rb1;
#pragma unroll
    for (int h = 0; h < RH; ++h) {
        const half_t* gA = (h == 0) ? gA0 : gA1;
        ra0[h] = *(const u32x4*)(gA);
        ra1[h] = *(const u32x4*)(gA + 8);
        *(u32x4*)&As[0][sr + h * 64][sk]     = ra0[h];
        *(u32x4*)&As[0][sr + h * 64][sk + 8] = ra1[h];
    }
    rb0 = *(const u32x4*)(gB);
    rb1 = *(const u32x4*)(gB + 8);
    *(u32x4*)&Bs[0][sr][sk]     = rb0;
    *(u32x4*)&Bs[0][sr][sk + 8] = rb1;
    __syncthreads();
#endif

    for (int kk = 0; kk < nk; ++kk) {
        const int buf = kk & 1;
#if USE_ASYNC
        if (kk + 1 < nk) {
            stage_tile((long)(kk + 1) * 32, buf ^ 1);   // lands during compute
            if (kk + 2 < nk) {
                __builtin_prefetch(gA0 + (long)(kk + 2) * 32, 0, 1);
                __builtin_prefetch(gB + (long)(kk + 2) * 32, 0, 1);
            }
        }
#else
        if (kk + 1 < nk) {
            const long ko = (long)(kk + 1) * 32;
#pragma unroll
            for (int h = 0; h < RH; ++h) {
                const half_t* nA = ((h == 0) ? gA0 : gA1) + ko;
                ra0[h] = *(const u32x4*)(nA);
                ra1[h] = *(const u32x4*)(nA + 8);
            }
            rb0 = *(const u32x4*)(gB + ko);
            rb1 = *(const u32x4*)(gB + ko + 8);
        }
#endif
        // ---- compute current tile ----
        {
            Frag af[MT], bf[2];
            const int rl = lane & 15;
            const int hi = lane >> 4;
#pragma unroll
            for (int mi = 0; mi < MT; ++mi) {
                const int m  = wm + mi * 16 + rl;
                const int kf = hi * 8;
                af[mi].u[0] = *(const u32x4*)&As[buf][m][kf];
                af[mi].u[1] = *(const u32x4*)&As[buf][m][16 + kf];
            }
#pragma unroll
            for (int ni = 0; ni < 2; ++ni) {
                const int n  = wn + ni * 16 + rl;
                const int kf = hi * 8;
                bf[ni].u[0] = *(const u32x4*)&Bs[buf][n][kf];
                bf[ni].u[1] = *(const u32x4*)&Bs[buf][n][16 + kf];
            }
#pragma unroll
            for (int mi = 0; mi < MT; ++mi)
#pragma unroll
                for (int ni = 0; ni < 2; ++ni)
                    acc[mi][ni] = __builtin_amdgcn_wmma_f32_16x16x32_f16(
                        false, af[mi].v, false, bf[ni].v,
                        (short)0, acc[mi][ni], false, false);
        }
#if USE_ASYNC
        if (kk + 1 < nk) __builtin_amdgcn_s_wait_asynccnt(0);
#else
        if (kk + 1 < nk) {
            const int nb = buf ^ 1;
#pragma unroll
            for (int h = 0; h < RH; ++h) {
                *(u32x4*)&As[nb][sr + h * 64][sk]     = ra0[h];
                *(u32x4*)&As[nb][sr + h * 64][sk + 8] = ra1[h];
            }
            *(u32x4*)&Bs[nb][sr][sk]     = rb0;
            *(u32x4*)&Bs[nb][sr][sk + 8] = rb1;
        }
#endif
        __syncthreads();
    }

    // ---- epilogue ----
#pragma unroll
    for (int mi = 0; mi < MT; ++mi) {
#pragma unroll
        for (int ni = 0; ni < 2; ++ni) {
            const int n  = n0 + wn + ni * 16 + (lane & 15);
            const int mb = m0 + wm + mi * 16 + (lane >> 4) * 8;
#pragma unroll
            for (int j = 0; j < 8; ++j) {
                const int m = mb + j;
                float v = acc[mi][ni][j] * alpha;
                if (bias2) v += bias2[m];
                if (scale) v = v * scale[m] + bias[m];
                if (resid) v += rtm ? resid[(long)n * ldr + m]
                                    : resid[(long)m * ldr + n];
                if (ct) C[(long)n * ldc + m] = v;
                else    C[(long)m * ldc + n] = v;
            }
        }
    }
}

// ---------------------------------------------------------------------------
// LIF over T=4. Buffers [T][B*CH*N], t-stride = B*CH*N. Layouts: TM [b][n][c]
// or NM [b][c][n]; thread enumerates OUTPUT layout, input remapped if needed.
// ---------------------------------------------------------------------------
template <bool IN_TM, bool OUT_TM>
__global__ void lif_kernel(const float* __restrict__ pre,
                           const float* __restrict__ pos,
                           half_t* __restrict__ spk,
                           long tstride, int CH, int N, long total)
{
    long i = (long)blockIdx.x * blockDim.x + threadIdx.x;
    if (i >= total) return;
    int c, n; long b;
    if (OUT_TM) { c = (int)(i % CH); long r = i / CH; n = (int)(r % N); b = r / N; }
    else        { n = (int)(i % N);  long r = i / N;  c = (int)(r % CH); b = r / CH; }
    long in_i;
    if (IN_TM == OUT_TM) in_i = i;
    else if (IN_TM)      in_i = (b * N + n) * CH + c;
    else                 in_i = (b * CH + c) * N + n;
    float p = 0.f;
    if (pos) p = pos[(long)c * N + n];
    float mem = 0.f, s = 0.f;
#pragma unroll
    for (int t = 0; t < 4; ++t) {
        const float u = pre[(long)t * tstride + in_i] + p;
        mem = mem * 0.25f * (1.f - s) + u;
        s = (mem - 0.5f) > 0.f ? 1.f : 0.f;
        spk[(long)t * tstride + i] = (half_t)s;
    }
}

// Depthwise 3x3 SAME on token-major f16 spikes [img][y][x][c]
__global__ void dwconv_kernel(const half_t* __restrict__ spk,
                              const float* __restrict__ w,
                              const float* __restrict__ s,
                              const float* __restrict__ b,
                              float* __restrict__ out,
                              int C, int H, int W, long total)
{
    long i = (long)blockIdx.x * blockDim.x + threadIdx.x;
    if (i >= total) return;
    const int c = (int)(i % C);
    long r = i / C;
    const int x = (int)(r % W);
    r /= W;
    const int y = (int)(r % H);
    const long img = r / H;
    const half_t* base = spk + img * (long)H * W * C;
    const float* wk = w + (long)c * 9;
    float acc = 0.f;
#pragma unroll
    for (int dy = 0; dy < 3; ++dy) {
        const int yy = y + dy - 1;
        if (yy < 0 || yy >= H) continue;
#pragma unroll
        for (int dx = 0; dx < 3; ++dx) {
            const int xx = x + dx - 1;
            if (xx < 0 || xx >= W) continue;
            acc += wk[dy * 3 + dx] * (float)base[((long)yy * W + xx) * C + c];
        }
    }
    out[i] = acc * s[c] + b[c];
}

__global__ void f2h_kernel(const float* __restrict__ in, half_t* __restrict__ out, long n)
{
    long i = (long)blockIdx.x * blockDim.x + threadIdx.x;
    if (i < n) out[i] = (half_t)in[i];
}

// ---------------------------------------------------------------------------
extern "C" void kernel_launch(void* const* d_in, const int* in_sizes, int n_in,
                              void* d_out, int out_size, void* d_ws, size_t ws_size,
                              hipStream_t stream)
{
    (void)in_sizes; (void)n_in; (void)out_size; (void)ws_size;

    const int T = 4, B = 8, C = 512, H = 24, W = 24;
    const int Nn = H * W, TB = T * B;
    const int med = 1024, Cv = 2048, hid = 2048;
    const int heads = 8, dh = 64, eh = 256;
    const long perT  = (long)B * C * Nn;
    const long perTm = (long)B * med * Nn;
    const long perTv = (long)B * Cv * Nn;

    const float* x      = (const float*)d_in[0];
    const float* pw1_w  = (const float*)d_in[1];
    const float* pw1_s  = (const float*)d_in[2];
    const float* pw1_b  = (const float*)d_in[3];
    const float* dw_w   = (const float*)d_in[4];
    const float* dw_s   = (const float*)d_in[5];
    const float* dw_b   = (const float*)d_in[6];
    const float* pw2_w  = (const float*)d_in[7];
    const float* pw2_s  = (const float*)d_in[8];
    const float* pw2_b  = (const float*)d_in[9];
    const float* s_pos  = (const float*)d_in[10];
    const float* q_w    = (const float*)d_in[11];
    const float* q_s    = (const float*)d_in[12];
    const float* q_b    = (const float*)d_in[13];
    const float* k_w    = (const float*)d_in[14];
    const float* k_s    = (const float*)d_in[15];
    const float* k_b    = (const float*)d_in[16];
    const float* v_w    = (const float*)d_in[17];
    const float* v_s    = (const float*)d_in[18];
    const float* v_b    = (const float*)d_in[19];
    const float* proj_w = (const float*)d_in[20];
    const float* proj_s = (const float*)d_in[21];
    const float* proj_b = (const float*)d_in[22];
    const float* fc1_w  = (const float*)d_in[23];
    const float* fc1_c  = (const float*)d_in[24];
    const float* fc1_s  = (const float*)d_in[25];
    const float* fc1_b  = (const float*)d_in[26];
    const float* fc2_w  = (const float*)d_in[27];
    const float* fc2_c  = (const float*)d_in[28];
    const float* fc2_s  = (const float*)d_in[29];
    const float* fc2_b  = (const float*)d_in[30];
    float* out = (float*)d_out;

    char* wsb = (char*)d_ws;
    size_t cur = 0;
    auto take = [&](size_t bytes) -> void* {
        cur = (cur + 255) & ~(size_t)255;
        void* p = wsb + cur;
        cur += bytes;
        return p;
    };
    const long TOK = (long)TB * Nn;

    half_t* Wpw1h  = (half_t*)take((size_t)med * C * 2);
    half_t* Wpw2h  = (half_t*)take((size_t)C * med * 2);
    half_t* Wqh    = (half_t*)take((size_t)C * C * 2);
    half_t* Wkh    = (half_t*)take((size_t)C * C * 2);
    half_t* Wvh    = (half_t*)take((size_t)Cv * C * 2);
    half_t* Wprojh = (half_t*)take((size_t)C * Cv * 2);
    half_t* Wfc1h  = (half_t*)take((size_t)hid * C * 2);
    half_t* Wfc2h  = (half_t*)take((size_t)C * hid * 2);
    half_t* SPK512  = (half_t*)take((size_t)C * TOK * 2);    // TM
    half_t* SPKQ    = (half_t*)take((size_t)C * TOK * 2);    // TM
    half_t* SPKK    = (half_t*)take((size_t)C * TOK * 2);    // NM
    half_t* SPKV    = (half_t*)take((size_t)Cv * TOK * 2);   // NM
    half_t* SPK1024 = (half_t*)take((size_t)med * TOK * 2);  // TM
    half_t* SPK2048 = (half_t*)take((size_t)Cv * TOK * 2);   // TM
    half_t* KVH     = (half_t*)take((size_t)TB * heads * dh * eh * 2);
    float* F512   = (float*)take((size_t)C * TOK * 4);
    float* F1024  = (float*)take((size_t)med * TOK * 4);
    float* F2048  = (float*)take((size_t)Cv * TOK * 4);
    float* KVF    = (float*)take((size_t)TB * heads * dh * eh * 4);
    float* XCUR   = (float*)take((size_t)C * TOK * 4);

    const int EB = 256;
    auto blocks = [&](long n) { return (unsigned)((n + EB - 1) / EB); };

    f2h_kernel<<<blocks((long)med * C), EB, 0, stream>>>(pw1_w,  Wpw1h,  (long)med * C);
    f2h_kernel<<<blocks((long)C * med), EB, 0, stream>>>(pw2_w,  Wpw2h,  (long)C * med);
    f2h_kernel<<<blocks((long)C * C),   EB, 0, stream>>>(q_w,    Wqh,    (long)C * C);
    f2h_kernel<<<blocks((long)C * C),   EB, 0, stream>>>(k_w,    Wkh,    (long)C * C);
    f2h_kernel<<<blocks((long)Cv * C),  EB, 0, stream>>>(v_w,    Wvh,    (long)Cv * C);
    f2h_kernel<<<blocks((long)C * Cv),  EB, 0, stream>>>(proj_w, Wprojh, (long)C * Cv);
    f2h_kernel<<<blocks((long)hid * C), EB, 0, stream>>>(fc1_w,  Wfc1h,  (long)hid * C);
    f2h_kernel<<<blocks((long)C * hid), EB, 0, stream>>>(fc2_w,  Wfc2h,  (long)C * hid);

    const long sT512  = (long)Nn * C;
    const long sT1024 = (long)Nn * med;
    const long sT2048 = (long)Nn * Cv;

    // =============== SepConv_Spike ===============
    lif_kernel<false, true><<<blocks(perT), EB, 0, stream>>>(
        x, nullptr, SPK512, perT, C, Nn, perT);
    wmma_gemm<128><<<dim3(Nn / 64, med / 128, TB), 128, 0, stream>>>(
        Wpw1h, 0, 0, C, SPK512, sT512, 0, C,
        F1024, sT1024, 0, med, 1,
        nullptr, pw1_s, pw1_b, nullptr, 0, 0, 0, 0,
        1, med, Nn, C, 1.0f);
    lif_kernel<true, true><<<blocks(perTm), EB, 0, stream>>>(
        F1024, nullptr, SPK1024, perTm, med, Nn, perTm);
    dwconv_kernel<<<blocks((long)TB * med * Nn), EB, 0, stream>>>(
        SPK1024, dw_w, dw_s, dw_b, F1024, med, H, W, (long)TB * med * Nn);
    lif_kernel<true, true><<<blocks(perTm), EB, 0, stream>>>(
        F1024, nullptr, SPK1024, perTm, med, Nn, perTm);
    wmma_gemm<128><<<dim3(Nn / 64, C / 128, TB), 128, 0, stream>>>(
        Wpw2h, 0, 0, med, SPK1024, sT1024, 0, med,
        XCUR, sT512, 0, C, 1,
        nullptr, pw2_s, pw2_b, x, (long)C * Nn, 0, Nn, 0,
        1, C, Nn, med, 1.0f);

    // =============== MS_Attention_linear ===============
    lif_kernel<true, true><<<blocks(perT), EB, 0, stream>>>(
        XCUR, s_pos, SPK512, perT, C, Nn, perT);
    wmma_gemm<128><<<dim3(Nn / 64, C / 128, TB), 128, 0, stream>>>(
        Wqh, 0, 0, C, SPK512, sT512, 0, C,
        F512, sT512, 0, C, 1,
        nullptr, q_s, q_b, nullptr, 0, 0, 0, 0,
        1, C, Nn, C, 1.0f);
    lif_kernel<true, true><<<blocks(perT), EB, 0, stream>>>(
        F512, nullptr, SPKQ, perT, C, Nn, perT);
    wmma_gemm<128><<<dim3(Nn / 64, C / 128, TB), 128, 0, stream>>>(
        Wkh, 0, 0, C, SPK512, sT512, 0, C,
        F512, sT512, 0, Nn, 0,
        nullptr, k_s, k_b, nullptr, 0, 0, 0, 0,
        1, C, Nn, C, 1.0f);
    lif_kernel<false, false><<<blocks(perT), EB, 0, stream>>>(
        F512, nullptr, SPKK, perT, C, Nn, perT);
    wmma_gemm<128><<<dim3(Nn / 64, Cv / 128, TB), 128, 0, stream>>>(
        Wvh, 0, 0, C, SPK512, sT512, 0, C,
        F2048, sT2048, 0, Nn, 0,
        nullptr, v_s, v_b, nullptr, 0, 0, 0, 0,
        1, Cv, Nn, C, 1.0f);
    lif_kernel<false, false><<<blocks(perTv), EB, 0, stream>>>(
        F2048, nullptr, SPKV, perTv, Cv, Nn, perTv);
    // kv[d,e] = sum_n k[d,n] v[e,n] -> stored transposed [e][d]
    wmma_gemm<64><<<dim3(eh / 64, dh / 64, TB * heads), 128, 0, stream>>>(
        SPKK, (long)C * Nn, (long)dh * Nn, Nn,
        SPKV, (long)Cv * Nn, (long)eh * Nn, Nn,
        KVF, (long)heads * dh * eh, (long)dh * eh, dh, 1,
        nullptr, nullptr, nullptr, nullptr, 0, 0, 0, 0,
        heads, dh, eh, Nn, 1.0f);
    f2h_kernel<<<blocks((long)TB * heads * dh * eh), EB, 0, stream>>>(
        KVF, KVH, (long)TB * heads * dh * eh);
    // o[e,n] = 0.25 * KV^T x q
    wmma_gemm<128><<<dim3(Nn / 64, eh / 128, TB * heads), 128, 0, stream>>>(
        KVH, (long)heads * dh * eh, (long)dh * eh, dh,
        SPKQ, sT512, (long)dh, C,
        F2048, sT2048, (long)eh, Cv, 1,
        nullptr, nullptr, nullptr, nullptr, 0, 0, 0, 0,
        heads, eh, Nn, dh, 0.25f);
    lif_kernel<true, true><<<blocks(perTv), EB, 0, stream>>>(
        F2048, nullptr, SPK2048, perTv, Cv, Nn, perTv);
    wmma_gemm<128><<<dim3(Nn / 64, C / 128, TB), 128, 0, stream>>>(
        Wprojh, 0, 0, Cv, SPK2048, sT2048, 0, Cv,
        XCUR, sT512, 0, C, 1,
        nullptr, proj_s, proj_b, XCUR, sT512, 0, C, 1,
        1, C, Nn, Cv, 1.0f);

    // =============== MS_MLP ===============
    lif_kernel<true, true><<<blocks(perT), EB, 0, stream>>>(
        XCUR, nullptr, SPK512, perT, C, Nn, perT);
    wmma_gemm<128><<<dim3(Nn / 64, hid / 128, TB), 128, 0, stream>>>(
        Wfc1h, 0, 0, C, SPK512, sT512, 0, C,
        F2048, sT2048, 0, hid, 1,
        fc1_c, fc1_s, fc1_b, nullptr, 0, 0, 0, 0,
        1, hid, Nn, C, 1.0f);
    lif_kernel<true, true><<<blocks(perTv), EB, 0, stream>>>(
        F2048, nullptr, SPK2048, perTv, hid, Nn, perTv);
    wmma_gemm<128><<<dim3(Nn / 64, C / 128, TB), 128, 0, stream>>>(
        Wfc2h, 0, 0, hid, SPK2048, sT2048, 0, hid,
        out, sT512, 0, Nn, 0,
        fc2_c, fc2_s, fc2_b, XCUR, sT512, 0, C, 1,
        1, C, Nn, hid, 1.0f);
}
